// Attention_54125177864799
// MI455X (gfx1250) — compile-verified, matrix operand order checked by reading, not run on hardware
//
#include <hip/hip_runtime.h>
#include <cstdint>

typedef __attribute__((ext_vector_type(16))) _Float16 v16h;
typedef __attribute__((ext_vector_type(8)))  _Float16 v8h;
typedef __attribute__((ext_vector_type(4)))  _Float16 v4h;
typedef __attribute__((ext_vector_type(8)))  float    v8f;
typedef __attribute__((ext_vector_type(4)))  unsigned int u32x4;
typedef __attribute__((ext_vector_type(8)))  int          i32x8;
typedef __attribute__((ext_vector_type(4)))  int          i32x4;

union FragU { v16h v; v8h h[2]; };

#define LDSP 72   // LDS row pitch in halves: 64-half row + 8-half pad (16B aligned)

// ---------------------------------------------------------------------------
// TDM: issue one 2D f16 tile load into LDS via the Tensor Data Mover.
// D# encoding per CDNA5 ISA ch.8: group0 = {count, lds_addr, global_addr, type},
// group1 = {flags/pad, tensor dims, tile dims, dim0 stride}. Pad config:
// interval code 4 (32 dwords = one 64-half row), amount code 3 (4 dwords =
// 8 halves) -> LDS rows land at pitch 72 halves, matching LDSP.
// Tracked by TENSORcnt; caller waits with s_wait_tensorcnt.
// ---------------------------------------------------------------------------
__device__ __forceinline__ void tdm_load_tile_f16(
    unsigned lds_byte_addr, const _Float16* gsrc,
    unsigned tile_d0 /*contiguous elems*/, unsigned tile_d1 /*rows*/,
    unsigned long long row_stride_elems)
{
    unsigned long long ga = (unsigned long long)(uintptr_t)gsrc;
    u32x4 g0;
    g0[0] = 1u;                                       // count=1 (valid user D#)
    g0[1] = lds_byte_addr;                            // lds_addr
    g0[2] = (unsigned)ga;                             // global_addr[31:0]
    g0[3] = (unsigned)((ga >> 32) & 0x01FFFFFFu)      // global_addr[56:32]
          | (2u << 30);                               // type = 2 ("image")
    const unsigned dim0 = 1u << 30, dim1 = 1u << 30;  // huge: no OOB clipping
    i32x8 g1;
    g1[0] = (int)((1u << 16)      // data_size = 1 (2 bytes)
          | (1u << 20)            // pad_enable
          | (4u << 22)            // pad_interval code 4 = 32 dwords
          | (3u << 25));          // pad_amount  code 3 = 4 dwords (8 halves)
    g1[1] = (int)((dim0 & 0xFFFFu) << 16);                    // tensor_dim0 lo16
    g1[2] = (int)((dim0 >> 16) | ((dim1 & 0xFFFFu) << 16));   // dim0 hi | dim1 lo
    g1[3] = (int)((dim1 >> 16) | (tile_d0 << 16));            // dim1 hi | tile_dim0
    g1[4] = (int)(tile_d1 & 0xFFFFu);                         // tile_dim1 (dim2=0)
    g1[5] = (int)(unsigned)(row_stride_elems & 0xFFFFFFFFull);// dim0_stride lo32
    g1[6] = (int)(unsigned)((row_stride_elems >> 32) & 0xFFFFull);
    g1[7] = 0;
    i32x4 z4 = {0, 0, 0, 0};
#if __clang_major__ >= 23
    i32x8 z8 = {0, 0, 0, 0, 0, 0, 0, 0};
    __builtin_amdgcn_tensor_load_to_lds(g0, g1, z4, z4, z8, 0);
#else
    __builtin_amdgcn_tensor_load_to_lds(g0, g1, z4, z4, 0);
#endif
}

// ---------------------------------------------------------------------------
// Batched GEMM: C[z] = alpha * A[z] @ Bt[z]^T (+ bias)
//   A : [M,K] row-major, f16 (TDM-staged) or f32 (converted during staging)
//   Bt: [N,K] row-major f16 (pre-transposed weights / K rows / V^T), TDM-staged
//   C : f32 or f16 per OUT_F16
// Block: 256 threads (8 waves), tile 128(M) x 64(N), K stage = 64, double-
// buffered LDS. Each wave owns a 32x32 sub-tile -> 4 WMMA accums, 8 WMMAs per
// stage. Batch z: b1 = z/nz2, b2 = z%nz2, offsets b1*s1 + b2*s2 (elements).
// ---------------------------------------------------------------------------
template <bool A_F32, bool OUT_F16>
__global__ __launch_bounds__(256) void gemm_wmma_kernel(
    const void* __restrict__ Av, long long sA1, long long sA2, int lda,
    const _Float16* __restrict__ Bt, long long sB1, long long sB2, int ldb,
    const float* __restrict__ bias,
    void* __restrict__ Cv, long long sC1, long long sC2, int ldc,
    int K, int nz2, float alpha)
{
    const int tid  = threadIdx.x;
    const int wave = tid >> 5;
    const int lane = tid & 31;
    const int l16  = lane & 15;
    const int hiK  = lane >> 4;

    const long long z  = blockIdx.z;
    const long long b1 = z / nz2;
    const long long b2 = z % nz2;
    const long long offA = b1 * sA1 + b2 * sA2;
    const _Float16* Ah = (const _Float16*)Av + offA;
    const float*    Af = (const float*)Av + offA;
    const _Float16* Bb = Bt + b1 * sB1 + b2 * sB2;
    const long long offC = b1 * sC1 + b2 * sC2;
    _Float16* Ch = (_Float16*)Cv + offC;
    float*    Cf = (float*)Cv + offC;

    const int m0    = blockIdx.y * 128;
    const int n0blk = blockIdx.x * 64;

    __shared__ alignas(16) _Float16 As[2][128 * LDSP];
    __shared__ alignas(16) _Float16 Bs[2][64 * LDSP];

    const int wm = (wave & 3) * 32;   // wave's 32-row slice
    const int wn = (wave >> 2) * 32;  // wave's 32-col slice

    v8f acc[2][2] = {{{}, {}}, {{}, {}}};

    auto issue_stage = [&](int kidx, int buf) {
        const int k0 = kidx * 64;
        if (wave == 0) {
            if (!A_F32)
                tdm_load_tile_f16((unsigned)(uintptr_t)&As[buf][0],
                                  Ah + (long long)m0 * lda + k0,
                                  64u, 128u, (unsigned long long)lda);
            tdm_load_tile_f16((unsigned)(uintptr_t)&Bs[buf][0],
                              Bb + (long long)n0blk * ldb + k0,
                              64u, 64u, (unsigned long long)ldb);
        }
        if (A_F32) {
            // synchronous f32 -> f16 staging of the 128x64 A tile
            const int r  = tid >> 1;
            const int c0 = (tid & 1) * 32;
            const float* src = Af + (long long)(m0 + r) * lda + k0 + c0;
            #pragma unroll
            for (int j = 0; j < 4; ++j) {
                float4 f0 = *(const float4*)(src + j * 8);
                float4 f1 = *(const float4*)(src + j * 8 + 4);
                v8h hv;
                hv[0] = (_Float16)f0.x; hv[1] = (_Float16)f0.y;
                hv[2] = (_Float16)f0.z; hv[3] = (_Float16)f0.w;
                hv[4] = (_Float16)f1.x; hv[5] = (_Float16)f1.y;
                hv[6] = (_Float16)f1.z; hv[7] = (_Float16)f1.w;
                *(v8h*)&As[buf][r * LDSP + c0 + j * 8] = hv;
            }
        }
    };

    issue_stage(0, 0);
    const int nStages = K >> 6;
    for (int i = 0; i < nStages; ++i) {
        const int buf = i & 1;
        if (wave == 0) __builtin_amdgcn_s_wait_tensorcnt(0);
        __syncthreads();                       // stage-i data in; buf^1 free
        if (i + 1 < nStages) issue_stage(i + 1, buf ^ 1);

        const _Float16* As_ = &As[buf][0];
        const _Float16* Bs_ = &Bs[buf][0];
        #pragma unroll
        for (int s = 0; s < 64; s += 32) {
            const int kb = s + hiK * 8;        // ISA f16 fragment K-mapping
            FragU a0, a1, fb0, fb1;
            a0.h[0]  = *(const v8h*)&As_[(wm + l16) * LDSP + kb];
            a0.h[1]  = *(const v8h*)&As_[(wm + l16) * LDSP + kb + 16];
            a1.h[0]  = *(const v8h*)&As_[(wm + 16 + l16) * LDSP + kb];
            a1.h[1]  = *(const v8h*)&As_[(wm + 16 + l16) * LDSP + kb + 16];
            fb0.h[0] = *(const v8h*)&Bs_[(wn + l16) * LDSP + kb];
            fb0.h[1] = *(const v8h*)&Bs_[(wn + l16) * LDSP + kb + 16];
            fb1.h[0] = *(const v8h*)&Bs_[(wn + 16 + l16) * LDSP + kb];
            fb1.h[1] = *(const v8h*)&Bs_[(wn + 16 + l16) * LDSP + kb + 16];

            acc[0][0] = __builtin_amdgcn_wmma_f32_16x16x32_f16(
                false, a0.v, false, fb0.v, (short)0, acc[0][0], false, false);
            acc[0][1] = __builtin_amdgcn_wmma_f32_16x16x32_f16(
                false, a0.v, false, fb1.v, (short)0, acc[0][1], false, false);
            acc[1][0] = __builtin_amdgcn_wmma_f32_16x16x32_f16(
                false, a1.v, false, fb0.v, (short)0, acc[1][0], false, false);
            acc[1][1] = __builtin_amdgcn_wmma_f32_16x16x32_f16(
                false, a1.v, false, fb1.v, (short)0, acc[1][1], false, false);
        }
        __syncthreads();                       // done reading buf before reuse
    }

    // epilogue: D layout lane = n%16 + 16*(m/8), vgpr = m%8
    #pragma unroll
    for (int tm = 0; tm < 2; ++tm) {
        #pragma unroll
        for (int tn = 0; tn < 2; ++tn) {
            const int nc = n0blk + wn + tn * 16 + l16;
            const float bv_ = bias ? bias[nc] : 0.0f;
            #pragma unroll
            for (int v = 0; v < 8; ++v) {
                const int mr = m0 + wm + tm * 16 + hiK * 8 + v;
                const float r = acc[tm][tn][v] * alpha + bv_;
                if (OUT_F16) Ch[(long long)mr * ldc + nc] = (_Float16)r;
                else         Cf[(long long)mr * ldc + nc] = r;
            }
        }
    }
}

// --------------------- elementwise f32 -> f16 conversion --------------------
__global__ __launch_bounds__(256) void cvt_f32_f16_kernel(
    const float* __restrict__ src, _Float16* __restrict__ dst)
{
    const long long i = ((long long)blockIdx.x * 256 + threadIdx.x) * 8;
    float4 f0 = *(const float4*)(src + i);
    float4 f1 = *(const float4*)(src + i + 4);
    v8h hv;
    hv[0] = (_Float16)f0.x; hv[1] = (_Float16)f0.y;
    hv[2] = (_Float16)f0.z; hv[3] = (_Float16)f0.w;
    hv[4] = (_Float16)f1.x; hv[5] = (_Float16)f1.y;
    hv[6] = (_Float16)f1.z; hv[7] = (_Float16)f1.w;
    *(v8h*)(dst + i) = hv;
}

// ------------- W [R,C] f32 -> Wt [C,R] f16 tiled transpose ------------------
__global__ __launch_bounds__(256) void transpose_cvt_kernel(
    const float* __restrict__ src, _Float16* __restrict__ dst, int R, int C)
{
    __shared__ _Float16 T[64 * 68];
    const int tid = threadIdx.x;
    const int r0 = blockIdx.y * 64, c0 = blockIdx.x * 64;
    const int rr = tid >> 4;         // 0..15
    const int cb = (tid & 15) * 4;   // 0..60
    #pragma unroll
    for (int q = 0; q < 4; ++q) {
        const int r = rr + q * 16;
        float4 f = *(const float4*)(src + (long long)(r0 + r) * C + c0 + cb);
        T[(cb + 0) * 68 + r] = (_Float16)f.x;
        T[(cb + 1) * 68 + r] = (_Float16)f.y;
        T[(cb + 2) * 68 + r] = (_Float16)f.z;
        T[(cb + 3) * 68 + r] = (_Float16)f.w;
    }
    __syncthreads();
    #pragma unroll
    for (int q = 0; q < 4; ++q) {
        const int r = rr + q * 16;   // out row within tile (= original col)
        v4h hv = *(const v4h*)&T[r * 68 + cb];
        *(v4h*)(dst + (long long)(c0 + r) * R + r0 + cb) = hv;
    }
}

// ---- Vh [B*N, D] f16 -> VhT [B][H][64][N] f16 (per-head transpose) ---------
__global__ __launch_bounds__(256) void transpose_head_kernel(
    const _Float16* __restrict__ Vh, _Float16* __restrict__ VhT)
{
    __shared__ _Float16 T[64 * 68];
    const int tid = threadIdx.x;
    const int zb  = blockIdx.z;
    const int b = zb >> 4, h = zb & 15;
    const int key0 = blockIdx.x * 64;
    const int rr = tid >> 4;
    const int cb = (tid & 15) * 4;
    const _Float16* src = Vh + (long long)b * (2048LL * 1024) +
                          (long long)key0 * 1024 + h * 64;
    #pragma unroll
    for (int q = 0; q < 4; ++q) {
        const int r = rr + q * 16;   // key within tile
        v4h hv = *(const v4h*)(src + (long long)r * 1024 + cb);
        T[(cb + 0) * 68 + r] = hv[0];
        T[(cb + 1) * 68 + r] = hv[1];
        T[(cb + 2) * 68 + r] = hv[2];
        T[(cb + 3) * 68 + r] = hv[3];
    }
    __syncthreads();
    _Float16* dst = VhT + (long long)b * (16LL * 64 * 2048) +
                    (long long)h * (64 * 2048);
    #pragma unroll
    for (int q = 0; q < 4; ++q) {
        const int d = rr + q * 16;
        v4h hv = *(const v4h*)&T[d * 68 + cb];
        *(v4h*)(dst + (long long)d * 2048 + key0 + cb) = hv;
    }
}

// ------------- in-place row softmax: one wave per 2048-wide row -------------
__global__ __launch_bounds__(256) void softmax_rows_kernel(float* __restrict__ attn)
{
    const int wave = threadIdx.x >> 5;
    const int lane = threadIdx.x & 31;
    const long long row = (long long)blockIdx.x * 8 + wave;
    float* p = attn + row * 2048;

    float4 v[16];
    #pragma unroll
    for (int i = 0; i < 16; ++i)
        v[i] = *(const float4*)(p + i * 128 + lane * 4);

    float m = -3.402823466e38f;
    #pragma unroll
    for (int i = 0; i < 16; ++i)
        m = fmaxf(m, fmaxf(fmaxf(v[i].x, v[i].y), fmaxf(v[i].z, v[i].w)));
    #pragma unroll
    for (int off = 16; off > 0; off >>= 1)
        m = fmaxf(m, __shfl_xor(m, off, 32));

    float s = 0.0f;
    #pragma unroll
    for (int i = 0; i < 16; ++i) {
        v[i].x = __expf(v[i].x - m); s += v[i].x;
        v[i].y = __expf(v[i].y - m); s += v[i].y;
        v[i].z = __expf(v[i].z - m); s += v[i].z;
        v[i].w = __expf(v[i].w - m); s += v[i].w;
    }
    #pragma unroll
    for (int off = 16; off > 0; off >>= 1)
        s += __shfl_xor(s, off, 32);

    const float inv = 1.0f / s;
    #pragma unroll
    for (int i = 0; i < 16; ++i) {
        v[i].x *= inv; v[i].y *= inv; v[i].z *= inv; v[i].w *= inv;
        *(float4*)(p + i * 128 + lane * 4) = v[i];
    }
}

extern "C" void kernel_launch(void* const* d_in, const int* in_sizes, int n_in,
                              void* d_out, int out_size, void* d_ws, size_t ws_size,
                              hipStream_t stream)
{
    (void)in_sizes; (void)n_in; (void)out_size; (void)ws_size;

    const long long SZ = 4194304LL;   // B*N*D
    const long long WZ = 1048576LL;   // D*D

    const float* x  = (const float*)d_in[0];
    const float* Wq = (const float*)d_in[1];
    const float* bq = (const float*)d_in[2];
    const float* Wk = (const float*)d_in[3];
    const float* bk = (const float*)d_in[4];
    const float* Wv = (const float*)d_in[5];
    const float* bv = (const float*)d_in[6];
    const float* Wo = (const float*)d_in[7];
    const float* bo = (const float*)d_in[8];

    // workspace layout (f16), total ~56 MB
    _Float16* Xh   = (_Float16*)d_ws;
    _Float16* Wqt  = Xh + SZ;
    _Float16* Wkt  = Wqt + WZ;
    _Float16* Wvt  = Wkt + WZ;
    _Float16* Wot  = Wvt + WZ;
    _Float16* Qh   = Wot + WZ;
    _Float16* Kh   = Qh + SZ;
    _Float16* Vh   = Kh + SZ;
    _Float16* VhT  = Vh + SZ;
    _Float16* Ctxh = VhT + SZ;

    float* out  = (float*)d_out;
    float* attn = out + SZ;

    const dim3 blk(256);

    // one-time f16 conversion / weight transposes
    cvt_f32_f16_kernel<<<dim3((unsigned)(SZ / 2048)), blk, 0, stream>>>(x, Xh);
    const dim3 tgrid(16, 16);
    transpose_cvt_kernel<<<tgrid, blk, 0, stream>>>(Wq, Wqt, 1024, 1024);
    transpose_cvt_kernel<<<tgrid, blk, 0, stream>>>(Wk, Wkt, 1024, 1024);
    transpose_cvt_kernel<<<tgrid, blk, 0, stream>>>(Wv, Wvt, 1024, 1024);
    transpose_cvt_kernel<<<tgrid, blk, 0, stream>>>(Wo, Wot, 1024, 1024);

    // QKV projections: [4096,1024] = Xh @ Wt^T + b  -> f16
    const dim3 gproj(16, 32, 1);
    gemm_wmma_kernel<false, true><<<gproj, blk, 0, stream>>>(
        Xh, 0, 0, 1024,  Wqt, 0, 0, 1024,  bq,  Qh, 0, 0, 1024,  1024, 1, 1.0f);
    gemm_wmma_kernel<false, true><<<gproj, blk, 0, stream>>>(
        Xh, 0, 0, 1024,  Wkt, 0, 0, 1024,  bk,  Kh, 0, 0, 1024,  1024, 1, 1.0f);
    gemm_wmma_kernel<false, true><<<gproj, blk, 0, stream>>>(
        Xh, 0, 0, 1024,  Wvt, 0, 0, 1024,  bv,  Vh, 0, 0, 1024,  1024, 1, 1.0f);

    // V -> per-head transposed [B][H][64][N]
    transpose_head_kernel<<<dim3(32, 1, 32), blk, 0, stream>>>(Vh, VhT);

    // scores: attn[b,h] = (Q_h @ K_h^T) / 8  -> f32 output region
    const long long sQb = 2097152LL, sQh = 64LL;
    const long long sCb = 67108864LL, sCh = 4194304LL;
    gemm_wmma_kernel<false, false><<<dim3(32, 16, 32), blk, 0, stream>>>(
        Qh, sQb, sQh, 1024,  Kh, sQb, sQh, 1024,  nullptr,
        attn, sCb, sCh, 2048,  64, 16, 0.125f);

    // softmax over the key axis, in place
    softmax_rows_kernel<<<dim3(8192), blk, 0, stream>>>(attn);

    // ctx[b,h] = attn[b,h] @ V_h  (A = f32 attn, B = VhT f16) -> f16
    gemm_wmma_kernel<true, true><<<dim3(1, 16, 32), blk, 0, stream>>>(
        attn, sCb, sCh, 2048,  VhT, 2097152LL, 131072LL, 2048,  nullptr,
        Ctxh, 2097152LL, 64LL, 1024,  2048, 16, 1.0f);

    // output projection: out = ctx @ Wo^T + bo -> f32
    gemm_wmma_kernel<false, false><<<gproj, blk, 0, stream>>>(
        Ctxh, 0, 0, 1024,  Wot, 0, 0, 1024,  bo,  out, 0, 0, 1024,  1024, 1, 1.0f);
}
